// TWINNET_LSTM_57028575756767
// MI455X (gfx1250) — matrix operand order repaired and checked by reading, not compiled, not run on hardware
//
#include <hip/hip_runtime.h>

#define B_      64
#define T_      512
#define H_      512
#define C_      64
#define NWGD    32      // workgroups per direction (each owns NJ h-columns)
#define NJ      16
#define THREADS 256     // 8 wave32s

typedef __attribute__((ext_vector_type(16))) __bf16    v16bf;
typedef __attribute__((ext_vector_type(8)))  float     v8f;
typedef __attribute__((ext_vector_type(4)))  unsigned  u32x4;

union Frag { u32x4 u[2]; v16bf v; };

// ---------------- workspace layout (bytes), per direction then shared ----------------
static constexpr size_t H1_OFF   = 0;                                  // [2][64][512] bf16 (ping-pong)
static constexpr size_t H2_OFF   = 131072;                             // [2][64][512] bf16
static constexpr size_t C1_OFF   = 262144;                             // [64][512]    bf16
static constexpr size_t YS_OFF   = 327680;                             // [T][64][512] bf16 (c2 sequence)
static constexpr size_t PER_DIR  = YS_OFF + (size_t)T_ * B_ * H_ * 2;  // 33,882,112
static constexpr size_t FCW_OFF  = 2 * PER_DIR;                        // [2][64][512] bf16 (fc_w, bfc_w)
static constexpr size_t SYNC_OFF = FCW_OFF + (size_t)2 * C_ * H_ * 2;  // 2 dirs x {cnt, gen}

// ---------------- LDS layout (bytes) ----------------
static constexpr unsigned SM_W1   = 0;        // [64][512] bf16 : w_hh (layer1) slice, [gate_col][k]
static constexpr unsigned SM_W2I  = 65536;    // [64][512] bf16 : w_ih (layer2) slice
static constexpr unsigned SM_W2H  = 131072;   // [64][512] bf16 : w_hh (layer2) slice
static constexpr unsigned SM_GATE = 196608;   // [64][64] f32 gate staging
static constexpr unsigned SM_C1   = 212992;   // [64][16] f32 cell state master (layer1)
static constexpr unsigned SM_C2   = 217088;   // [64][16] f32 cell state master (layer2)
static constexpr unsigned SM_B1   = 221184;   // [64] f32 fused bias layer1
static constexpr unsigned SM_B2   = 221440;   // [64] f32 fused bias layer2
static constexpr unsigned SM_WIH1 = 221696;   // [64] f32 w_ih1 column (input size 1)
static constexpr unsigned SM_SIZE = 221952;   // < 320KB WGP LDS

#define WMMA_BF16(a, b, c) \
  __builtin_amdgcn_wmma_f32_16x16x32_bf16(false, (a), false, (b), (short)0, (c), false, false)

__device__ __forceinline__ float sigmoidf_(float x) { return 1.0f / (1.0f + __expf(-x)); }

// A-fragment (16x32 bf16) from a [rows][512] bf16 row-major matrix.
// ISA 16-bit A layout: lanes 0-15 row M=lane, K chunks {0..7, 16..23};
// lanes 16-31 same rows, K chunks {8..15, 24..31}. Two contiguous 16B loads.
__device__ __forceinline__ v16bf a_frag(const __bf16* m, int mt, int kk) {
  int lane = threadIdx.x & 31;
  int r  = mt * 16 + (lane & 15);
  int k0 = kk * 32 + ((lane & 16) ? 8 : 0);
  Frag f;
  f.u[0] = *(const u32x4*)(m + (size_t)r * H_ + k0);
  f.u[1] = *(const u32x4*)(m + (size_t)r * H_ + k0 + 16);
  return f.v;
}

// B-fragment (32x16 bf16) from a [col][512] bf16 matrix (col = N index).
// 16-bit B layout (matches a_frag): lanes 0-15 col N=lane, K {0..7, 16..23}; lanes 16-31 K {8..15, 24..31}.
// two runs of 8 contiguous bf16 per lane (k0, k0+16) = two 16B loads.
__device__ __forceinline__ v16bf b_frag(const __bf16* w, int nt, int kk) {
  int lane = threadIdx.x & 31;
  int c  = nt * 16 + (lane & 15);
  int k0 = kk * 32 + ((lane & 16) ? 8 : 0);
  Frag f;
  f.u[0] = *(const u32x4*)(w + (size_t)c * H_ + k0);
  f.u[1] = *(const u32x4*)(w + (size_t)c * H_ + k0 + 16);
  return f.v;
}

// Scatter a 16x16 f32 accumulator tile into LDS gate staging [64][64].
__device__ __forceinline__ void store_acc(float* gates, int mt, int nt, v8f acc) {
  int lane = threadIdx.x & 31;
  int n  = nt * 16 + (lane & 15);
  int r0 = mt * 16 + ((lane & 16) ? 8 : 0);
#pragma unroll
  for (int v = 0; v < 8; ++v) gates[(r0 + v) * 64 + n] = acc[v];
}

// Direction-local grid barrier (32 WGs), sense counter + generation, s_sleep spin.
__device__ __forceinline__ void dir_sync(unsigned* cnt, unsigned* gen) {
  __syncthreads();
  if (threadIdx.x == 0) {
    __threadfence();                                   // release h/c stores
    unsigned g = *(volatile unsigned*)gen;
    if (atomicAdd(cnt, 1u) == NWGD - 1) {
      *(volatile unsigned*)cnt = 0;
      __threadfence();
      atomicAdd(gen, 1u);
    } else {
      while (*(volatile unsigned*)gen == g) __builtin_amdgcn_s_sleep(1);
    }
    __threadfence();                                   // acquire
  }
  __syncthreads();
}

// --------------------------------------------------------------------------
// Prep: zero sync counters, zero ping-pong h buffers (read at t=0),
// convert fc weights to bf16.
// --------------------------------------------------------------------------
__global__ void prep_kernel(const float* __restrict__ fcw, const float* __restrict__ bfcw,
                            char* __restrict__ ws) {
  unsigned tid = blockIdx.x * blockDim.x + threadIdx.x;
  if (tid < 8) ((unsigned*)(ws + SYNC_OFF))[tid] = 0u;
  if (tid < 2u * C_ * H_) {
    __bf16* fw = (__bf16*)(ws + FCW_OFF);
    const float* src = (tid < (unsigned)(C_ * H_)) ? fcw : bfcw;
    fw[tid] = (__bf16)src[tid & (C_ * H_ - 1)];
  }
  if (tid < 4u * B_ * H_) {                 // 2 dirs x {h1,h2} x [64][512], buffer 0 only
    unsigned d   = tid >> 16;
    unsigned rem = tid & 65535u;
    unsigned mat = rem >> 15;
    unsigned e   = rem & 32767u;
    __bf16* p = (__bf16*)(ws + d * PER_DIR + (mat ? H2_OFF : H1_OFF));
    p[e] = (__bf16)0.0f;
  }
}

// --------------------------------------------------------------------------
// Persistent twin-LSTM kernel: 64 WGs = 2 directions x 32 WGs.
// Recurrent weights live in LDS for all 512 steps; h exchanged via L2.
// --------------------------------------------------------------------------
__global__ __launch_bounds__(THREADS, 1)
void lstm_persistent(const float* __restrict__ x,
                     const float* w_ih1, const float* w_hh1, const float* b_ih1, const float* b_hh1,
                     const float* w_ih2, const float* w_hh2, const float* b_ih2, const float* b_hh2,
                     const float* w_ih3, const float* w_hh3, const float* b_ih3, const float* b_hh3,
                     const float* w_ih4, const float* w_hh4, const float* b_ih4, const float* b_hh4,
                     char* __restrict__ ws) {
  extern __shared__ char smem[];
  __bf16* W1    = (__bf16*)(smem + SM_W1);
  __bf16* W2i   = (__bf16*)(smem + SM_W2I);
  __bf16* W2h   = (__bf16*)(smem + SM_W2H);
  float*  gates = (float*)(smem + SM_GATE);
  float*  c1s   = (float*)(smem + SM_C1);
  float*  c2s   = (float*)(smem + SM_C2);
  float*  bias1 = (float*)(smem + SM_B1);
  float*  bias2 = (float*)(smem + SM_B2);
  float*  wih1  = (float*)(smem + SM_WIH1);

  const int dir = blockIdx.x >> 5;     // 0 = fwd, 1 = bwd
  const int wg  = blockIdx.x & 31;     // owns h columns [wg*NJ, wg*NJ+NJ)
  const int wv  = threadIdx.x >> 5;

  const float *l1_wih, *l1_whh, *l1_bih, *l1_bhh, *l2_wih, *l2_whh, *l2_bih, *l2_bhh;
  if (dir == 0) { l1_wih = w_ih1; l1_whh = w_hh1; l1_bih = b_ih1; l1_bhh = b_hh1;
                  l2_wih = w_ih2; l2_whh = w_hh2; l2_bih = b_ih2; l2_bhh = b_hh2; }
  else          { l1_wih = w_ih3; l1_whh = w_hh3; l1_bih = b_ih3; l1_bhh = b_hh3;
                  l2_wih = w_ih4; l2_whh = w_hh4; l2_bih = b_ih4; l2_bhh = b_hh4; }

  // ---- stage weight slices into LDS (f32 -> bf16), [gate_col][k] ----
  for (int idx = threadIdx.x; idx < 64 * H_; idx += THREADS) {
    int l = idx >> 9, k = idx & (H_ - 1);
    int g = (l >> 4) * H_ + wg * NJ + (l & 15);          // global gate row
    W1[idx]  = (__bf16)l1_whh[(size_t)g * H_ + k];
    W2i[idx] = (__bf16)l2_wih[(size_t)g * H_ + k];
    W2h[idx] = (__bf16)l2_whh[(size_t)g * H_ + k];
  }
  if (threadIdx.x < 64) {
    int l = threadIdx.x;
    int g = (l >> 4) * H_ + wg * NJ + (l & 15);
    bias1[l] = l1_bih[g] + l1_bhh[g];
    bias2[l] = l2_bih[g] + l2_bhh[g];
    wih1[l]  = l1_wih[g];                                // w_ih1 is (4H, 1)
  }
  for (int e = threadIdx.x; e < B_ * NJ; e += THREADS) { c1s[e] = 0.0f; c2s[e] = 0.0f; }
  __syncthreads();   // h buffers zeroed by prep_kernel (stream-ordered)

  char* dws = ws + (size_t)dir * PER_DIR;
  __bf16* h1  = (__bf16*)(dws + H1_OFF);
  __bf16* h2  = (__bf16*)(dws + H2_OFF);
  __bf16* c1g = (__bf16*)(dws + C1_OFF);
  __bf16* ys  = (__bf16*)(dws + YS_OFF);
  unsigned* cnt = (unsigned*)(ws + SYNC_OFF) + dir * 2;
  unsigned* gen = cnt + 1;

  for (int t = 0; t < T_; ++t) {
    const __bf16* h1r = h1 + (size_t)(t & 1) * (B_ * H_);
    __bf16*       h1w = h1 + (size_t)((t + 1) & 1) * (B_ * H_);
    const __bf16* h2r = h2 + (size_t)(t & 1) * (B_ * H_);
    __bf16*       h2w = h2 + (size_t)((t + 1) & 1) * (B_ * H_);

    __builtin_prefetch(h2r + threadIdx.x * 64, 0, 0);    // warm layer-2 operand

    // ===== layer 1: gates = h1_prev @ W1slice  (16 tiles, 2 per wave) =====
#pragma unroll
    for (int u = 0; u < 2; ++u) {
      int ti = 2 * wv + u, mt = ti >> 2, nt = ti & 3;
      v8f acc = {};
#pragma unroll 4
      for (int kk = 0; kk < H_ / 32; ++kk)
        acc = WMMA_BF16(a_frag(h1r, mt, kk), b_frag(W1, nt, kk), acc);
      store_acc(gates, mt, nt, acc);
    }
    __syncthreads();

    // ===== layer 1 activations (x term folded here; input size is 1) =====
    {
      int tt = (dir == 0) ? t : (T_ - 1 - t);
      for (int e = threadIdx.x; e < B_ * NJ; e += THREADS) {
        int b = e >> 4, jj = e & 15;
        float xb = x[(size_t)b * T_ + tt];
        float gi = gates[b * 64 + jj]      + bias1[jj]      + xb * wih1[jj];
        float gf = gates[b * 64 + 16 + jj] + bias1[16 + jj] + xb * wih1[16 + jj];
        float gg = gates[b * 64 + 32 + jj] + bias1[32 + jj] + xb * wih1[32 + jj];
        float go = gates[b * 64 + 48 + jj] + bias1[48 + jj] + xb * wih1[48 + jj];
        float cn = sigmoidf_(gf) * c1s[e] + sigmoidf_(gi) * tanhf(gg);
        float hn = sigmoidf_(go) * tanhf(cn);
        c1s[e] = cn;
        int col = wg * NJ + jj;
        h1w[(size_t)b * H_ + col] = (__bf16)hn;
        c1g[(size_t)b * H_ + col] = (__bf16)cn;          // layer-2 input is c1
      }
    }
    dir_sync(cnt, gen);

    // ===== layer 2: gates = c1 @ W2i + h2_prev @ W2h =====
#pragma unroll
    for (int u = 0; u < 2; ++u) {
      int ti = 2 * wv + u, mt = ti >> 2, nt = ti & 3;
      v8f acc = {};
#pragma unroll 4
      for (int kk = 0; kk < H_ / 32; ++kk)
        acc = WMMA_BF16(a_frag(c1g, mt, kk), b_frag(W2i, nt, kk), acc);
#pragma unroll 4
      for (int kk = 0; kk < H_ / 32; ++kk)
        acc = WMMA_BF16(a_frag(h2r, mt, kk), b_frag(W2h, nt, kk), acc);
      store_acc(gates, mt, nt, acc);
    }
    __syncthreads();

    // ===== layer 2 activations; c2 is the per-step output =====
    for (int e = threadIdx.x; e < B_ * NJ; e += THREADS) {
      int b = e >> 4, jj = e & 15;
      float gi = gates[b * 64 + jj]      + bias2[jj];
      float gf = gates[b * 64 + 16 + jj] + bias2[16 + jj];
      float gg = gates[b * 64 + 32 + jj] + bias2[32 + jj];
      float go = gates[b * 64 + 48 + jj] + bias2[48 + jj];
      float cn = sigmoidf_(gf) * c2s[e] + sigmoidf_(gi) * tanhf(gg);
      float hn = sigmoidf_(go) * tanhf(cn);
      c2s[e] = cn;
      int col = wg * NJ + jj;
      h2w[(size_t)b * H_ + col] = (__bf16)hn;
      ys[((size_t)t * B_ + b) * H_ + col] = (__bf16)cn;  // bwd stays in reversed order (matches ref)
    }
    dir_sync(cnt, gen);
  }
}

// --------------------------------------------------------------------------
// Final FC: out[dir][b][t][c] = ys[t*64+b][:] . fc_w[c][:] + fc_b[c]
// M = T*B = 32768, N = C = 64, K = H = 512. 64 rows per WG, 2 tiles/wave.
// --------------------------------------------------------------------------
__global__ __launch_bounds__(THREADS, 2)
void fc_kernel(const char* __restrict__ ws, const float* __restrict__ fcb0,
               const float* __restrict__ fcb1, float* __restrict__ out) {
  int dir = blockIdx.x >> 9;          // 512 blocks per direction
  int rb  = blockIdx.x & 511;
  const __bf16* ys = (const __bf16*)(ws + (size_t)dir * PER_DIR + YS_OFF);
  const __bf16* fw = (const __bf16*)(ws + FCW_OFF + (size_t)dir * C_ * H_ * 2);
  const float*  fb = dir ? fcb1 : fcb0;
  float* o = out + (size_t)dir * B_ * T_ * C_;

  int wv = threadIdx.x >> 5, lane = threadIdx.x & 31;
  int r0 = rb * 64;
  const __bf16* a_base = ys + (size_t)r0 * H_;
  __builtin_prefetch(a_base + threadIdx.x * 128, 0, 0);

#pragma unroll
  for (int u = 0; u < 2; ++u) {
    int ti = 2 * wv + u, mt = ti >> 2, nt = ti & 3;
    v8f acc = {};
#pragma unroll 4
    for (int kk = 0; kk < H_ / 32; ++kk)
      acc = WMMA_BF16(a_frag(a_base, mt, kk), b_frag(fw, nt, kk), acc);
    int c  = nt * 16 + (lane & 15);
    float bv = fb[c];
    int rr = r0 + mt * 16 + ((lane & 16) ? 8 : 0);
#pragma unroll
    for (int v = 0; v < 8; ++v) {
      int r = rr + v, t = r >> 6, b = r & 63;
      o[(size_t)b * (T_ * C_) + (size_t)t * C_ + c] = acc[v] + bv;
    }
  }
}

extern "C" void kernel_launch(void* const* d_in, const int* in_sizes, int n_in,
                              void* d_out, int out_size, void* d_ws, size_t ws_size,
                              hipStream_t stream) {
  const float* x     = (const float*)d_in[0];
  const float* wih1  = (const float*)d_in[1];
  const float* whh1  = (const float*)d_in[2];
  const float* bih1  = (const float*)d_in[3];
  const float* bhh1  = (const float*)d_in[4];
  const float* wih2  = (const float*)d_in[5];
  const float* whh2  = (const float*)d_in[6];
  const float* bih2  = (const float*)d_in[7];
  const float* bhh2  = (const float*)d_in[8];
  const float* wih3  = (const float*)d_in[9];
  const float* whh3  = (const float*)d_in[10];
  const float* bih3  = (const float*)d_in[11];
  const float* bhh3  = (const float*)d_in[12];
  const float* wih4  = (const float*)d_in[13];
  const float* whh4  = (const float*)d_in[14];
  const float* bih4  = (const float*)d_in[15];
  const float* bhh4  = (const float*)d_in[16];
  const float* fcw   = (const float*)d_in[17];
  const float* fcb   = (const float*)d_in[18];
  const float* bfcw  = (const float*)d_in[19];
  const float* bfcb  = (const float*)d_in[20];
  char* ws = (char*)d_ws;

  prep_kernel<<<dim3(512), dim3(256), 0, stream>>>(fcw, bfcw, ws);

  lstm_persistent<<<dim3(2 * NWGD), dim3(THREADS), SM_SIZE, stream>>>(
      x, wih1, whh1, bih1, bhh1, wih2, whh2, bih2, bhh2,
      wih3, whh3, bih3, bhh3, wih4, whh4, bih4, bhh4, ws);

  fc_kernel<<<dim3(2 * 512), dim3(THREADS), 0, stream>>>(ws, fcb, bfcb, (float*)d_out);
}
